// DROIDNetwork_62036507623702
// MI455X (gfx1250) — matrix operand order repaired and checked by reading, not compiled, not run on hardware
//
#include <hip/hip_runtime.h>
#include <hip/hip_bf16.h>

typedef __attribute__((ext_vector_type(16))) _Float16 v16h;
typedef __attribute__((ext_vector_type(8)))  float    v8f;

// ---------------------------------------------------------------- conversions
__global__ __launch_bounds__(256) void cvt_f32_to_f16(const float* __restrict__ x,
                                                      _Float16* __restrict__ y, long n) {
  long i = (long)blockIdx.x * blockDim.x + threadIdx.x;
  if (i < n) y[i] = (_Float16)x[i];
}

// pack OIHW f32 weights -> f16 [O][Kp], K-order = (c,kh,kw), zero-pad to Kp
__global__ __launch_bounds__(256) void pack_weights(const float* __restrict__ w,
                                                    _Float16* __restrict__ wp,
                                                    int O, int Kdim, int Kp) {
  long i = (long)blockIdx.x * blockDim.x + threadIdx.x;
  if (i >= (long)O * Kp) return;
  int o = (int)(i / Kp), k = (int)(i % Kp);
  wp[i] = (k < Kdim) ? (_Float16)w[(long)o * Kdim + k] : (_Float16)0.f;
}

// pack OIHW f32 weights -> f16 [O][Kp], K-order = (kh,kw,c)  (Kp == Kdim)
__global__ __launch_bounds__(256) void pack_weights_hwc(const float* __restrict__ w,
                                                        _Float16* __restrict__ wp,
                                                        int O, int Cin, int KK) {
  long total = (long)O * Cin * KK;
  long i = (long)blockIdx.x * blockDim.x + threadIdx.x;
  if (i >= total) return;
  int Kp = Cin * KK;
  int o = (int)(i / Kp), k = (int)(i % Kp);
  int pos = k / Cin, c = k - pos * Cin;
  wp[i] = (_Float16)w[((long)o * Cin + c) * KK + pos];
}

// ------------------------------------------------- implicit-GEMM conv (WMMA)
// Wave tile: 16(M) x 64(N) -> 4 accumulators, one shared A gather + 4
// contiguous-32B B loads per 32-wide K-step (v_wmma_f32_16x16x32_f16).
// A (16x32 f16): lane L: M=L&15, half=L>>4,
//   element j -> K-offset = half*16 + j   (contiguous, same order as B)
// B (32x16 f16): lane L: col=L&15, element j -> K = half*16 + j  == one
//   contiguous 16-half load from packed [O][Kp] weights.
// FAST path (CIN%32==0, 3x3 layers): K packed as (kh,kw,c) so a whole K-step
// shares one (kh,kw): ONE bounds check per lane per step, and the 16 A
// elements sit at base + {0..15}*H*W -> immediate-offset loads.
template <int CIN, int H, int W, int OH, int OW, int KH, int STRIDE, int PAD,
          int KDIM, int KP, int O>
__global__ __launch_bounds__(256) void conv_wmma_f16(
    const _Float16* __restrict__ in, const _Float16* __restrict__ wpack,
    const float* __restrict__ bias, float* __restrict__ out) {
  constexpr int KK  = KH * KH;
  constexpr int HWo = OH * OW;
  constexpr int HWi = H * W;
  constexpr int oGroups = O >> 6;
  constexpr bool FAST = (CIN % 32 == 0);   // implies KP == KDIM == KK*CIN

  const int wave = threadIdx.x >> 5;
  const int lane = threadIdx.x & 31;
  const int half = lane >> 4;
  const int l16  = lane & 15;
  const int tile = blockIdx.x * 8 + wave;
  const int og = tile % oGroups;
  const int mt = tile / oGroups;

  // fixed output row for this lane's A fragment
  const int mrow = (mt << 4) + l16;
  const int n   = mrow / HWo;
  const int ohw = mrow % HWo;
  const int oh  = ohw / OW;
  const int ow  = ohw % OW;
  const int ih0 = oh * STRIDE - PAD;
  const int iw0 = ow * STRIDE - PAD;
  const _Float16* __restrict__ inN = in + (long)n * CIN * HWi;

  const _Float16* __restrict__ bp0 =
      wpack + (long)((og << 6) + l16) * KP + (half << 4);
  constexpr long bstep = (long)16 * KP;     // next 16-channel fragment

  v8f acc0 = {}, acc1 = {}, acc2 = {}, acc3 = {};
  for (int k0 = 0; k0 < KP; k0 += 32) {
    v16h b0 = *(const v16h*)(bp0 + k0);
    v16h b1 = *(const v16h*)(bp0 + bstep + k0);
    v16h b2 = *(const v16h*)(bp0 + 2 * bstep + k0);
    v16h b3 = *(const v16h*)(bp0 + 3 * bstep + k0);
    union { v16h v; _Float16 h[16]; } af;

    if constexpr (FAST) {
      // whole step shares one filter tap
      const int pos = k0 / CIN;             // shift (CIN power of two)
      const int kh  = pos / KH;             // const-div by 3
      const int kw  = pos - kh * KH;
      const int cb  = (k0 & (CIN - 1)) + (half << 4);
      const int ih = ih0 + kh, iw = iw0 + kw;
      if ((unsigned)ih < (unsigned)H && (unsigned)iw < (unsigned)W) {
        const _Float16* __restrict__ src = inN + cb * HWi + ih * W + iw;
#pragma unroll
        for (int j = 0; j < 16; ++j) {
          constexpr int co_tbl = 0;  (void)co_tbl;
          const int co = j;   // {0..15}
          af.h[j] = src[co * HWi];                    // immediate offsets
        }
      } else {
#pragma unroll
        for (int j = 0; j < 16; ++j) af.h[j] = (_Float16)0.f;
      }
    } else {
      // generic path (7x7, tiny Cin): per-element decompose, const divs
#pragma unroll
      for (int j = 0; j < 16; ++j) {
        int k = k0 + (half << 4) + j;
        _Float16 val = (_Float16)0.f;
        if (k < KDIM) {
          int c  = k / KK;
          int r  = k - c * KK;
          int kh = r / KH;
          int kw = r - kh * KH;
          int ih = ih0 + kh, iw = iw0 + kw;
          if ((unsigned)ih < (unsigned)H && (unsigned)iw < (unsigned)W)
            val = inN[c * HWi + ih * W + iw];
        }
        af.h[j] = val;
      }
    }

    acc0 = __builtin_amdgcn_wmma_f32_16x16x32_f16(false, af.v, false, b0,
                                                  (short)0, acc0, false, false);
    acc1 = __builtin_amdgcn_wmma_f32_16x16x32_f16(false, af.v, false, b1,
                                                  (short)0, acc1, false, false);
    acc2 = __builtin_amdgcn_wmma_f32_16x16x32_f16(false, af.v, false, b2,
                                                  (short)0, acc2, false, false);
    acc3 = __builtin_amdgcn_wmma_f32_16x16x32_f16(false, af.v, false, b3,
                                                  (short)0, acc3, false, false);
  }

  // D layout: lane L -> col=L&15, rows r+(L>>4)*8; rows are consecutive output
  // pixels (same n, channel) -> two float4 stores per fragment.
  const int m0   = (mt << 4) + (half << 3);
  const int n2   = m0 / HWo;
  const int phw0 = m0 % HWo;
  v8f accs[4] = {acc0, acc1, acc2, acc3};
#pragma unroll
  for (int f = 0; f < 4; ++f) {
    int ocol = (og << 6) + (f << 4) + l16;
    float bv = bias[ocol];
    union { v8f v; float4 q[2]; } u;
#pragma unroll
    for (int r = 0; r < 8; ++r) u.v[r] = accs[f][r] + bv;
    float* dst = out + ((long)n2 * O + ocol) * HWo + phw0;
    *(float4*)(dst)     = u.q[0];
    *(float4*)(dst + 4) = u.q[1];
  }
}

// --------------------------------------------------------------- group norm
__global__ __launch_bounds__(256) void gn_stats(const float* __restrict__ x,
                                                float* __restrict__ stats,
                                                int Cg, int HW) {
  __shared__ float ss[256], sq[256];
  const long cnt  = (long)Cg * HW;
  const long base = (long)blockIdx.x * cnt;   // groups contiguous in NCHW
  float s = 0.f, q = 0.f;
  for (long i = threadIdx.x; i < cnt; i += 256) {
    float v = x[base + i]; s += v; q += v * v;
  }
  ss[threadIdx.x] = s; sq[threadIdx.x] = q;
  __syncthreads();
  for (int o = 128; o > 0; o >>= 1) {
    if (threadIdx.x < o) { ss[threadIdx.x] += ss[threadIdx.x + o];
                           sq[threadIdx.x] += sq[threadIdx.x + o]; }
    __syncthreads();
  }
  if (threadIdx.x == 0) {
    float mu  = ss[0] / (float)cnt;
    float var = sq[0] / (float)cnt - mu * mu;
    stats[2 * blockIdx.x]     = mu;
    stats[2 * blockIdx.x + 1] = rsqrtf(var + 1e-5f);
  }
}

__global__ __launch_bounds__(256) void gn_apply_relu(
    const float* __restrict__ x, const float* __restrict__ stats,
    const float* __restrict__ gamma, const float* __restrict__ beta,
    _Float16* __restrict__ y, long total, int C, int HW, int Cg, int G) {
  long i = (long)blockIdx.x * blockDim.x + threadIdx.x;
  if (i >= total) return;
  int c = (int)((i / HW) % C);
  int n = (int)(i / ((long)C * HW));
  int g = c / Cg;
  float mu = stats[2 * (n * G + g)], rs = stats[2 * (n * G + g) + 1];
  float v  = (x[i] - mu) * rs * gamma[c] + beta[c];
  y[i] = (_Float16)fmaxf(v, 0.f);
}

// ------------------------------------------------------------- feature means
__global__ __launch_bounds__(256) void feat_mean(const _Float16* __restrict__ x3,
                                                 const _Float16* __restrict__ d2,
                                                 float* __restrict__ feat, int HW) {
  __shared__ float ss[256];
  int n = blockIdx.x, c = blockIdx.y;          // c in [0,256)
  const _Float16* src = (c < 128) ? x3 + ((long)n * 128 + c) * HW
                                  : d2 + ((long)n * 128 + (c - 128)) * HW;
  float s = 0.f;
  for (int i = threadIdx.x; i < HW; i += 256) s += (float)src[i];
  ss[threadIdx.x] = s;
  __syncthreads();
  for (int o = 128; o > 0; o >>= 1) {
    if (threadIdx.x < o) ss[threadIdx.x] += ss[threadIdx.x + o];
    __syncthreads();
  }
  if (threadIdx.x == 0) feat[n * 256 + c] = ss[0] / (float)HW;
}

// ------------------------------------------------------- GRU + SE(3) updates
__device__ __forceinline__ float sigm(float x) { return 1.f / (1.f + expf(-x)); }

__global__ __launch_bounds__(256) void gru_pose(
    const float* __restrict__ feat,
    const float* __restrict__ wih, const float* __restrict__ whh,
    const float* __restrict__ bih, const float* __restrict__ bhh,
    const float* __restrict__ pd1w, const float* __restrict__ pd1b,
    const float* __restrict__ pd2w, const float* __restrict__ pd2b,
    const float* __restrict__ cf1w, const float* __restrict__ cf1b,
    const float* __restrict__ cf2w, const float* __restrict__ cf2b,
    const int* __restrict__ numit,
    float* __restrict__ gi, float* __restrict__ hr1, float* __restrict__ cfr,
    float* __restrict__ outp) {
  __shared__ float h[64 * 128];
  __shared__ float poses[64 * 7];
  const int tid = threadIdx.x;
  for (int i = tid; i < 64 * 128; i += 256) h[i] = 0.f;
  for (int i = tid; i < 64 * 7; i += 256) poses[i] = (i % 7 == 0) ? 1.f : 0.f;
  __syncthreads();

  // gi depends only on features -> compute once
  for (int i = tid; i < 64 * 384; i += 256) {
    int n = i / 384, j = i % 384;
    const float* f = feat + n * 256;
    const float* w = wih + (long)j * 256;
    float s = bih[j];
    for (int k = 0; k < 256; ++k) s += f[k] * w[k];
    gi[i] = s;
  }
  __syncthreads();

  const int iters = *numit;
  for (int it = 0; it < iters; ++it) {
    // ---- GRU hidden update
    float hnew[32];
    int cnt = 0;
    for (int i = tid; i < 64 * 128; i += 256, ++cnt) {
      int n = i >> 7, j = i & 127;
      const float* hrow = h + (n << 7);
      const float* w0 = whh + (long)j * 128;
      const float* w1 = whh + (long)(j + 128) * 128;
      const float* w2 = whh + (long)(j + 256) * 128;
      float s0 = bhh[j], s1 = bhh[j + 128], s2 = bhh[j + 256];
      for (int k = 0; k < 128; ++k) {
        float hv = hrow[k];
        s0 += w0[k] * hv; s1 += w1[k] * hv; s2 += w2[k] * hv;
      }
      float r = sigm(gi[n * 384 + j] + s0);
      float z = sigm(gi[n * 384 + 128 + j] + s1);
      float nn = tanhf(gi[n * 384 + 256 + j] + r * s2);
      hnew[cnt] = (1.f - z) * nn + z * hrow[j];
    }
    __syncthreads();
    cnt = 0;
    for (int i = tid; i < 64 * 128; i += 256, ++cnt) h[i] = hnew[cnt];
    __syncthreads();

    // ---- pose-delta hidden & confidence hidden
    for (int i = tid; i < 64 * 128; i += 256) {
      int n = i >> 7, j = i & 127;
      const float* hrow = h + (n << 7);
      const float* w = pd1w + (long)j * 128;
      float s = pd1b[j];
      for (int k = 0; k < 128; ++k) s += hrow[k] * w[k];
      hr1[i] = fmaxf(s, 0.f);
    }
    for (int i = tid; i < 64 * 64; i += 256) {
      int n = i >> 6, j = i & 63;
      const float* hrow = h + (n << 7);
      const float* w = cf1w + (long)j * 128;
      float s = cf1b[j];
      for (int k = 0; k < 128; ++k) s += hrow[k] * w[k];
      cfr[i] = fmaxf(s, 0.f);
    }
    __syncthreads();

    // ---- per-pose SE(3) update (one thread per pose)
    if (tid < 64) {
      const int n = tid;
      float s = cf2b[0];
      for (int k = 0; k < 64; ++k) s += cfr[n * 64 + k] * cf2w[k];
      const float wgt = sigm(s);
      float xi[6];
      for (int kk = 0; kk < 6; ++kk) {
        float d = pd2b[kk];
        const float* w = pd2w + kk * 128;
        for (int k = 0; k < 128; ++k) d += hr1[n * 128 + k] * w[k];
        xi[kk] = d * wgt * 0.1f;
      }
      // exp_se3
      float v0 = xi[0], v1 = xi[1], v2 = xi[2];
      float w0 = xi[3], w1 = xi[4], w2 = xi[5];
      float th2 = w0 * w0 + w1 * w1 + w2 * w2;
      bool small = th2 < 1e-8f;
      float th2s = small ? 1.f : th2;
      float th = sqrtf(th2s);
      float A = small ? (1.f - th2 / 6.f) : (sinf(th) / th);
      float B = small ? (0.5f - th2 / 24.f) : ((1.f - cosf(th)) / th2s);
      float Cc = small ? (1.f / 6.f - th2 / 120.f) : ((1.f - A) / th2s);
      float Wm[9] = {0.f, -w2, w1, w2, 0.f, -w0, -w1, w0, 0.f};
      float W2[9];
      for (int i2 = 0; i2 < 3; ++i2)
        for (int j2 = 0; j2 < 3; ++j2) {
          float acc = 0.f;
          for (int k2 = 0; k2 < 3; ++k2) acc += Wm[i2 * 3 + k2] * Wm[k2 * 3 + j2];
          W2[i2 * 3 + j2] = acc;
        }
      float Re[9], Ve[9], te[3];
      for (int e = 0; e < 9; ++e) {
        float id = (e % 4 == 0) ? 1.f : 0.f;
        Re[e] = id + A * Wm[e] + B * W2[e];
        Ve[e] = id + B * Wm[e] + Cc * W2[e];
      }
      te[0] = Ve[0] * v0 + Ve[1] * v1 + Ve[2] * v2;
      te[1] = Ve[3] * v0 + Ve[4] * v1 + Ve[5] * v2;
      te[2] = Ve[6] * v0 + Ve[7] * v1 + Ve[8] * v2;
      // quat_to_T of current pose
      float qw = poses[n * 7 + 0], qx = poses[n * 7 + 1];
      float qy = poses[n * 7 + 2], qz = poses[n * 7 + 3];
      float qn2 = rsqrtf(qw * qw + qx * qx + qy * qy + qz * qz);
      qw *= qn2; qx *= qn2; qy *= qn2; qz *= qn2;
      float Rq[9] = {
        1.f - 2.f * (qy * qy + qz * qz), 2.f * (qx * qy - qw * qz), 2.f * (qx * qz + qw * qy),
        2.f * (qx * qy + qw * qz), 1.f - 2.f * (qx * qx + qz * qz), 2.f * (qy * qz - qw * qx),
        2.f * (qx * qz - qw * qy), 2.f * (qy * qz + qw * qx), 1.f - 2.f * (qx * qx + qy * qy)};
      float tq[3] = {poses[n * 7 + 4], poses[n * 7 + 5], poses[n * 7 + 6]};
      // compose T_new = exp @ T(pose)
      float Rn[9], tn[3];
      for (int i2 = 0; i2 < 3; ++i2) {
        for (int j2 = 0; j2 < 3; ++j2) {
          float acc = 0.f;
          for (int k2 = 0; k2 < 3; ++k2) acc += Re[i2 * 3 + k2] * Rq[k2 * 3 + j2];
          Rn[i2 * 3 + j2] = acc;
        }
        tn[i2] = Re[i2 * 3 + 0] * tq[0] + Re[i2 * 3 + 1] * tq[1] +
                 Re[i2 * 3 + 2] * tq[2] + te[i2];
      }
      // T_to_pose
      float m00 = Rn[0], m01 = Rn[1], m02 = Rn[2];
      float m10 = Rn[3], m11 = Rn[4], m12 = Rn[5];
      float m20 = Rn[6], m21 = Rn[7], m22 = Rn[8];
      float sv[4] = {1.f + m00 + m11 + m22, 1.f + m00 - m11 - m22,
                     1.f - m00 + m11 - m22, 1.f - m00 - m11 + m22};
      float qa[4];
      for (int e = 0; e < 4; ++e) qa[e] = sqrtf(fmaxf(sv[e], 1e-12f));
      float cand[4][4] = {
        {qa[0] * qa[0], m21 - m12, m02 - m20, m10 - m01},
        {m21 - m12, qa[1] * qa[1], m10 + m01, m02 + m20},
        {m02 - m20, m10 + m01, qa[2] * qa[2], m12 + m21},
        {m10 - m01, m20 + m02, m21 + m12, qa[3] * qa[3]}};
      int idx = 0;
      for (int e = 1; e < 4; ++e) if (qa[e] > qa[idx]) idx = e;
      float inv = 1.f / (2.f * fmaxf(qa[idx], 0.1f));
      float q0 = cand[idx][0] * inv, q1 = cand[idx][1] * inv;
      float q2 = cand[idx][2] * inv, q3 = cand[idx][3] * inv;
      float rn = rsqrtf(q0 * q0 + q1 * q1 + q2 * q2 + q3 * q3);
      if (n > 0) {   // mask: pose 0 stays identity
        poses[n * 7 + 0] = q0 * rn; poses[n * 7 + 1] = q1 * rn;
        poses[n * 7 + 2] = q2 * rn; poses[n * 7 + 3] = q3 * rn;
        poses[n * 7 + 4] = tn[0]; poses[n * 7 + 5] = tn[1]; poses[n * 7 + 6] = tn[2];
      }
    }
    __syncthreads();
  }
  for (int i = tid; i < 64 * 7; i += 256) outp[i] = poses[i];
}

// --------------------------------------------------------------------- host
extern "C" void kernel_launch(void* const* d_in, const int* in_sizes, int n_in,
                              void* d_out, int out_size, void* d_ws, size_t ws_size,
                              hipStream_t stream) {
  (void)in_sizes; (void)n_in; (void)out_size; (void)ws_size;
  const float* images = (const float*)d_in[0];
  const float* depths = (const float*)d_in[1];
  const float* ic1w = (const float*)d_in[2];  const float* ic1b = (const float*)d_in[3];
  const float* ig1g = (const float*)d_in[4];  const float* ig1b = (const float*)d_in[5];
  const float* ic2w = (const float*)d_in[6];  const float* ic2b = (const float*)d_in[7];
  const float* ig2g = (const float*)d_in[8];  const float* ig2b = (const float*)d_in[9];
  const float* ic3w = (const float*)d_in[10]; const float* ic3b = (const float*)d_in[11];
  const float* ig3g = (const float*)d_in[12]; const float* ig3b = (const float*)d_in[13];
  const float* dc1w = (const float*)d_in[14]; const float* dc1b = (const float*)d_in[15];
  const float* dg1g = (const float*)d_in[16]; const float* dg1b = (const float*)d_in[17];
  const float* dc2w = (const float*)d_in[18]; const float* dc2b = (const float*)d_in[19];
  const float* dg2g = (const float*)d_in[20]; const float* dg2b = (const float*)d_in[21];
  const float* gwih = (const float*)d_in[22]; const float* gwhh = (const float*)d_in[23];
  const float* gbih = (const float*)d_in[24]; const float* gbhh = (const float*)d_in[25];
  const float* pd1w = (const float*)d_in[26]; const float* pd1b = (const float*)d_in[27];
  const float* pd2w = (const float*)d_in[28]; const float* pd2b = (const float*)d_in[29];
  const float* cf1w = (const float*)d_in[30]; const float* cf1b = (const float*)d_in[31];
  const float* cf2w = (const float*)d_in[32]; const float* cf2b = (const float*)d_in[33];
  const int*   numit = (const int*)d_in[34];
  float* outp = (float*)d_out;

  // ---- workspace carving (256B aligned bump allocator)
  char* p = (char*)d_ws;
  auto alloc = [&](size_t bytes) -> void* {
    void* r = (void*)p;
    p += (bytes + 255) & ~(size_t)255;
    return r;
  };
  _Float16* imgH = (_Float16*)alloc((size_t)64 * 3 * 65536 * 2);
  _Float16* depH = (_Float16*)alloc((size_t)64 * 65536 * 2);
  _Float16* wp1  = (_Float16*)alloc((size_t)64 * 160 * 2);
  _Float16* wp2  = (_Float16*)alloc((size_t)128 * 576 * 2);
  _Float16* wp3  = (_Float16*)alloc((size_t)128 * 1152 * 2);
  _Float16* wpd1 = (_Float16*)alloc((size_t)64 * 64 * 2);
  _Float16* wpd2 = (_Float16*)alloc((size_t)128 * 576 * 2);
  float*    convf = (float*)alloc((size_t)64 * 64 * 16384 * 4);   // max conv out
  _Float16* bufA = (_Float16*)alloc((size_t)64 * 64 * 16384 * 2); // x1 / d1
  _Float16* bufB = (_Float16*)alloc((size_t)64 * 128 * 4096 * 2); // x2 / d2
  _Float16* bufC = (_Float16*)alloc((size_t)64 * 128 * 4096 * 2); // x3
  float*    stats = (float*)alloc((size_t)64 * 8 * 2 * 4);
  float*    feat  = (float*)alloc((size_t)64 * 256 * 4);
  float*    giB   = (float*)alloc((size_t)64 * 384 * 4);
  float*    hr1B  = (float*)alloc((size_t)64 * 128 * 4);
  float*    cfrB  = (float*)alloc((size_t)64 * 64 * 4);

  // ---- precision conversion + weight packing
  {
    long n = (long)64 * 3 * 65536;
    cvt_f32_to_f16<<<dim3((unsigned)((n + 255) / 256)), 256, 0, stream>>>(images, imgH, n);
    n = (long)64 * 65536;
    cvt_f32_to_f16<<<dim3((unsigned)((n + 255) / 256)), 256, 0, stream>>>(depths, depH, n);
  }
  // 7x7 layers: (c,kh,kw) order, padded
  pack_weights<<<dim3((64 * 160 + 255) / 256), 256, 0, stream>>>(ic1w, wp1, 64, 147, 160);
  pack_weights<<<dim3((64 * 64 + 255) / 256), 256, 0, stream>>>(dc1w, wpd1, 64, 49, 64);
  // 3x3 layers: (kh,kw,c) order (FAST path)
  pack_weights_hwc<<<dim3((128 * 576 + 255) / 256), 256, 0, stream>>>(ic2w, wp2, 128, 64, 9);
  pack_weights_hwc<<<dim3((128 * 1152 + 255) / 256), 256, 0, stream>>>(ic3w, wp3, 128, 128, 9);
  pack_weights_hwc<<<dim3((128 * 576 + 255) / 256), 256, 0, stream>>>(dc2w, wpd2, 128, 64, 9);

  auto run_gn = [&](const float* x, const float* g, const float* b, _Float16* y,
                    int C, int HW) {
    int Cg = C / 8;
    long total = (long)64 * C * HW;
    gn_stats<<<dim3(64 * 8), 256, 0, stream>>>(x, stats, Cg, HW);
    gn_apply_relu<<<dim3((unsigned)((total + 255) / 256)), 256, 0, stream>>>(
        x, stats, g, b, y, total, C, HW, Cg, 8);
  };

  // ---- image branch
  // ic1: (64,3,256,256)->(64,64,128,128)  M/16=65536, O/64=1 -> 8192 blocks
  conv_wmma_f16<3, 256, 256, 128, 128, 7, 2, 3, 147, 160, 64>
      <<<dim3(8192), 256, 0, stream>>>(imgH, wp1, ic1b, convf);
  run_gn(convf, ig1g, ig1b, bufA, 64, 16384);
  // ic2: (64,64,128,128)->(64,128,64,64)  M/16=16384, O/64=2 -> 4096 blocks
  conv_wmma_f16<64, 128, 128, 64, 64, 3, 2, 1, 576, 576, 128>
      <<<dim3(4096), 256, 0, stream>>>(bufA, wp2, ic2b, convf);
  run_gn(convf, ig2g, ig2b, bufB, 128, 4096);
  // ic3: (64,128,64,64)->(64,128,64,64)
  conv_wmma_f16<128, 64, 64, 64, 64, 3, 1, 1, 1152, 1152, 128>
      <<<dim3(4096), 256, 0, stream>>>(bufB, wp3, ic3b, convf);
  run_gn(convf, ig3g, ig3b, bufC, 128, 4096);

  // ---- depth branch (reuses bufA/bufB, now dead)
  conv_wmma_f16<1, 256, 256, 128, 128, 7, 2, 3, 49, 64, 64>
      <<<dim3(8192), 256, 0, stream>>>(depH, wpd1, dc1b, convf);
  run_gn(convf, dg1g, dg1b, bufA, 64, 16384);
  conv_wmma_f16<64, 128, 128, 64, 64, 3, 2, 1, 576, 576, 128>
      <<<dim3(4096), 256, 0, stream>>>(bufA, wpd2, dc2b, convf);
  run_gn(convf, dg2g, dg2b, bufB, 128, 4096);

  // ---- features + GRU/pose
  feat_mean<<<dim3(64, 256), 256, 0, stream>>>(bufC, bufB, feat, 4096);
  gru_pose<<<dim3(1), 256, 0, stream>>>(feat, gwih, gwhh, gbih, gbhh,
      pd1w, pd1b, pd2w, pd2b, cf1w, cf1b, cf2w, cf2b, numit,
      giB, hr1B, cfrB, outp);
}
